// GCN_44633300140578
// MI455X (gfx1250) — matrix operand order repaired and checked by reading, not compiled, not run on hardware
//
#include <hip/hip_runtime.h>
#include <hip/hip_bf16.h>

#define N_NODES   100000
#define N_EDGES   1600000
#define HID       128
#define N_CLASSES 40
#define N_LAYERS  3
#define LN_EPS    1e-5f

typedef __attribute__((ext_vector_type(16))) __bf16 v16bf;
typedef __attribute__((ext_vector_type(8)))  __bf16 v8bf;
typedef __attribute__((ext_vector_type(8)))  float  v8f;

// LDS row stride (in bf16 elements): 136 -> 272 B rows (16B aligned for b128),
// dword stride 68 == 4 (mod 64) -> conflict-free b128 fragment reads.
#define LSTR 136
#define ROWS_PER_BLOCK 32

__device__ __forceinline__ __bf16 f2bf(float f) {
    return static_cast<__bf16>(f);   // native v_cvt (RNE) on gfx1250
}
__device__ __forceinline__ unsigned pack2bf(float lo, float hi) {
    unsigned short a = __builtin_bit_cast(unsigned short, f2bf(lo));
    unsigned short b = __builtin_bit_cast(unsigned short, f2bf(hi));
    return (unsigned)a | ((unsigned)b << 16);
}

// ---------------------------------------------------------------------------
// One-time: convert all layer weights to transposed bf16: Wt[l][n*128+k]
// ---------------------------------------------------------------------------
__global__ void convert_w(const float* __restrict__ Ws, __bf16* __restrict__ Wt) {
    int idx = blockIdx.x * blockDim.x + threadIdx.x;       // 3*16384 total
    if (idx >= N_LAYERS * HID * HID) return;
    int layer = idx >> 14;
    int local = idx & 16383;
    int n = local >> 7;
    int k = local & 127;
    Wt[(size_t)layer * HID * HID + n * HID + k] =
        f2bf(Ws[(size_t)layer * HID * HID + k * HID + n]);
}

// ---------------------------------------------------------------------------
// Dense projection Y = X @ W (X:[N,128] f32, Wt:[128,128] bf16 transposed).
// Block = 256 threads = 8 waves; block covers 32 rows x 128 cols.
// Wave w owns columns [16w,16w+16) over 2 M-tiles; K=128 -> 4 K-steps.
// Per K-step: B fragment loaded once, reused by 2 v_wmma_f32_16x16x32_bf16.
// A tile (32x128) converted to bf16 in LDS once per block; W tile copied
// from preconverted global bf16. All fragments read with ds_load_b128.
// ---------------------------------------------------------------------------
__global__ __launch_bounds__(256) void gemm128_wmma(
    const float* __restrict__ X, const __bf16* __restrict__ Wt,
    float* __restrict__ Y)
{
    __shared__ __bf16 sW[HID * LSTR];              // sW[n][k]
    __shared__ __bf16 sX[ROWS_PER_BLOCK * LSTR];   // sX[r][k]

    const int tid = threadIdx.x;
    const int rowBase = blockIdx.x * ROWS_PER_BLOCK;

    // Stage W: bf16 global -> LDS via dword copies (coalesced).
    {
        const unsigned* wsrc = reinterpret_cast<const unsigned*>(Wt);
        unsigned* wdst = reinterpret_cast<unsigned*>(sW);
        for (int idx = tid; idx < HID * 64; idx += 256) {   // 64 dwords per row
            int n = idx >> 6;
            int kk = idx & 63;
            wdst[n * (LSTR / 2) + kk] = wsrc[n * 64 + kk];
        }
    }
    // Stage X tile: f32 global -> bf16 LDS (convert once per block).
    {
        unsigned* xdst = reinterpret_cast<unsigned*>(sX);
        for (int idx = tid; idx < ROWS_PER_BLOCK * 64; idx += 256) {
            int r = idx >> 6;
            int cc = idx & 63;
            const float2 v = *reinterpret_cast<const float2*>(
                X + (size_t)(rowBase + r) * HID + cc * 2);
            xdst[r * (LSTR / 2) + cc] = pack2bf(v.x, v.y);
        }
    }
    __syncthreads();

    const int wave = tid >> 5;          // 0..7 -> column tile
    const int lane = tid & 31;
    const int nBase = wave * 16;
    const int m   = lane & 15;          // A row / B column within tile
    const int klo = (lane >> 4) * 8;    // A K-phase per half-wave
    const int kbB = (lane >> 4) * 16;   // unused: B is read at klo, like A

    v8f acc0 = {};
    v8f acc1 = {};
#pragma unroll
    for (int kt = 0; kt < 4; ++kt) {
        const int kbase = kt * 32;

        // B fragment: column nBase+m, K = kbase+klo+{0..7, 16..23} (reused 2x)
        const v8bf blo = *reinterpret_cast<const v8bf*>(&sW[(nBase + m) * LSTR + kbase + klo]);
        const v8bf bhi = *reinterpret_cast<const v8bf*>(&sW[(nBase + m) * LSTR + kbase + klo + 16]);
        const v16bf b = __builtin_shufflevector(blo, bhi,
            0,1,2,3,4,5,6,7,8,9,10,11,12,13,14,15);

        // A fragment, M-tile 0 (rows 0..15)
        const v8bf a0lo = *reinterpret_cast<const v8bf*>(&sX[m * LSTR + kbase + klo]);
        const v8bf a0hi = *reinterpret_cast<const v8bf*>(&sX[m * LSTR + kbase + klo + 16]);
        const v16bf a0 = __builtin_shufflevector(a0lo, a0hi,
            0,1,2,3,4,5,6,7,8,9,10,11,12,13,14,15);

        // A fragment, M-tile 1 (rows 16..31)
        const v8bf a1lo = *reinterpret_cast<const v8bf*>(&sX[(16 + m) * LSTR + kbase + klo]);
        const v8bf a1hi = *reinterpret_cast<const v8bf*>(&sX[(16 + m) * LSTR + kbase + klo + 16]);
        const v16bf a1 = __builtin_shufflevector(a1lo, a1hi,
            0,1,2,3,4,5,6,7,8,9,10,11,12,13,14,15);

        acc0 = __builtin_amdgcn_wmma_f32_16x16x32_bf16(
            false, a0, false, b, (short)0, acc0, false, false);
        acc1 = __builtin_amdgcn_wmma_f32_16x16x32_bf16(
            false, a1, false, b, (short)0, acc1, false, false);
    }

    // C/D layout: element j -> M = j + (lane>=16 ? 8 : 0), N = lane&15
    const int mhalf = (lane >> 4) * 8;
    const int ncol = nBase + m;
#pragma unroll
    for (int j = 0; j < 8; ++j) {
        Y[(size_t)(rowBase + mhalf + j) * HID + ncol] = acc0[j];
        Y[(size_t)(rowBase + 16 + mhalf + j) * HID + ncol] = acc1[j];
    }
}

// ---------------------------------------------------------------------------
// Zero-fill
// ---------------------------------------------------------------------------
__global__ void zero_f32(float* __restrict__ p, int n) {
    int i = blockIdx.x * blockDim.x + threadIdx.x;
    int stride = gridDim.x * blockDim.x;
    for (; i < n; i += stride) p[i] = 0.0f;
}

// ---------------------------------------------------------------------------
// SpMM scatter-add: out[row[e], :] += src[col[e], :]
// One wave per edge, float4 per lane, f32 atomics (L2-resident destination).
// ---------------------------------------------------------------------------
__global__ __launch_bounds__(256) void spmm_scatter(
    const float* __restrict__ src, const int* __restrict__ erow,
    const int* __restrict__ ecol, float* __restrict__ out, int nE)
{
    const int lane = threadIdx.x & 31;
    const int gw = blockIdx.x * 8 + (threadIdx.x >> 5);
    const int nW = gridDim.x * 8;
    for (int e = gw; e < nE; e += nW) {
        const int r = erow[e];           // destination row (wave-uniform)
        const int c = ecol[e];           // source row
        const float4 v = *reinterpret_cast<const float4*>(src + (size_t)c * HID + lane * 4);
        float* o = out + (size_t)r * HID + lane * 4;
        atomicAdd(o + 0, v.x);
        atomicAdd(o + 1, v.y);
        atomicAdd(o + 2, v.z);
        atomicAdd(o + 3, v.w);
    }
}

// ---------------------------------------------------------------------------
// LayerNorm + ReLU, in place. One wave per node (wave32 shuffle reduction).
// ---------------------------------------------------------------------------
__global__ __launch_bounds__(256) void ln_relu(
    float* __restrict__ h, const float* __restrict__ scale,
    const float* __restrict__ bias)
{
    const int lane = threadIdx.x & 31;
    const int node = blockIdx.x * 8 + (threadIdx.x >> 5);
    float* row = h + (size_t)node * HID;

    float4 v = reinterpret_cast<const float4*>(row)[lane];
    float s  = v.x + v.y + v.z + v.w;
    float s2 = v.x * v.x + v.y * v.y + v.z * v.z + v.w * v.w;
#pragma unroll
    for (int off = 16; off > 0; off >>= 1) {
        s  += __shfl_xor(s, off, 32);
        s2 += __shfl_xor(s2, off, 32);
    }
    const float mu  = s * (1.0f / 128.0f);
    const float var = s2 * (1.0f / 128.0f) - mu * mu;
    const float inv = rsqrtf(var + LN_EPS);

    const float4 sc = reinterpret_cast<const float4*>(scale)[lane];
    const float4 bs = reinterpret_cast<const float4*>(bias)[lane];
    float4 o;
    o.x = fmaxf(0.0f, (v.x - mu) * inv * sc.x + bs.x);
    o.y = fmaxf(0.0f, (v.y - mu) * inv * sc.y + bs.y);
    o.z = fmaxf(0.0f, (v.z - mu) * inv * sc.z + bs.z);
    o.w = fmaxf(0.0f, (v.w - mu) * inv * sc.w + bs.w);
    reinterpret_cast<float4*>(row)[lane] = o;
}

// ---------------------------------------------------------------------------
// Head: logits = h @ W_out + b_out, then log_softmax. One block per node.
// ---------------------------------------------------------------------------
__global__ __launch_bounds__(128) void head_logsoftmax(
    const float* __restrict__ h, const float* __restrict__ Wo,
    const float* __restrict__ bo, float* __restrict__ out)
{
    __shared__ float sh[HID];
    __shared__ float sl[N_CLASSES];
    __shared__ float sred[2];

    const int node = blockIdx.x;
    const int t = threadIdx.x;

    sh[t] = h[(size_t)node * HID + t];
    __syncthreads();

    if (t < N_CLASSES) {
        float acc = bo[t];
#pragma unroll 8
        for (int k = 0; k < HID; ++k) acc += sh[k] * Wo[k * N_CLASSES + t];
        sl[t] = acc;
    }
    __syncthreads();

    if (t == 0) {
        float m = sl[0];
        for (int c = 1; c < N_CLASSES; ++c) m = fmaxf(m, sl[c]);
        float se = 0.0f;
        for (int c = 0; c < N_CLASSES; ++c) se += expf(sl[c] - m);
        sred[0] = m;
        sred[1] = logf(se);
    }
    __syncthreads();

    if (t < N_CLASSES)
        out[(size_t)node * N_CLASSES + t] = sl[t] - sred[0] - sred[1];
}

// ---------------------------------------------------------------------------
// Orchestration
// ---------------------------------------------------------------------------
extern "C" void kernel_launch(void* const* d_in, const int* in_sizes, int n_in,
                              void* d_out, int out_size, void* d_ws, size_t ws_size,
                              hipStream_t stream) {
    const float* x    = (const float*)d_in[0];
    const int*   ei   = (const int*)d_in[1];
    const int*   erow = ei;             // edge_index[0] = destination rows
    const int*   ecol = ei + N_EDGES;   // edge_index[1] = gathered source rows
    const float* Ws   = (const float*)d_in[2];
    const float* lns  = (const float*)d_in[3];
    const float* lnb  = (const float*)d_in[4];
    const float* Wo   = (const float*)d_in[5];
    const float* bo   = (const float*)d_in[6];
    float* out = (float*)d_out;

    float* buf0 = (float*)d_ws;                       // GEMM output
    float* buf1 = buf0 + (size_t)N_NODES * HID;       // aggregation output
    __bf16* wt  = (__bf16*)(buf1 + (size_t)N_NODES * HID);  // 3x transposed bf16 W

    convert_w<<<(N_LAYERS * HID * HID + 255) / 256, 256, 0, stream>>>(Ws, wt);

    const float* src = x;
    for (int i = 0; i < N_LAYERS; ++i) {
        gemm128_wmma<<<N_NODES / ROWS_PER_BLOCK, 256, 0, stream>>>(
            src, wt + (size_t)i * HID * HID, buf0);
        zero_f32<<<8192, 256, 0, stream>>>(buf1, N_NODES * HID);
        spmm_scatter<<<4000, 256, 0, stream>>>(buf0, erow, ecol, buf1, N_EDGES);
        ln_relu<<<N_NODES / 8, 256, 0, stream>>>(buf1, lns + i * HID, lnb + i * HID);
        src = buf1;
    }
    head_logsoftmax<<<N_NODES, 128, 0, stream>>>(buf1, Wo, bo, out);
}
